// NeuronAttentionBase_49014166782301
// MI455X (gfx1250) — matrix operand order repaired and checked by reading, not compiled, not run on hardware
//
#include <hip/hip_runtime.h>
#include <hip/hip_bf16.h>

typedef __bf16 bf16_t;
typedef __attribute__((ext_vector_type(16))) __bf16 v16bf;
typedef __attribute__((ext_vector_type(8)))  __bf16 v8bf;
typedef __attribute__((ext_vector_type(4)))  __bf16 v4bf;
typedef __attribute__((ext_vector_type(8)))  float  v8f;
typedef __attribute__((ext_vector_type(4)))  float  v4f;
typedef __attribute__((ext_vector_type(4)))  unsigned int u32x4;
typedef __attribute__((ext_vector_type(8)))  unsigned int u32x8;

#define WMMA_BF16(a, b, c) \
  __builtin_amdgcn_wmma_f32_16x16x32_bf16(false, (a), false, (b), (short)0, (c), false, false)

static __device__ __forceinline__ v16bf cat8(v8bf lo, v8bf hi) {
  return __builtin_shufflevector(lo, hi, 0,1,2,3,4,5,6,7,8,9,10,11,12,13,14,15);
}

static __device__ __forceinline__ float rowmax16(float v) {
  v = fmaxf(v, __shfl_xor(v, 1, 16));
  v = fmaxf(v, __shfl_xor(v, 2, 16));
  v = fmaxf(v, __shfl_xor(v, 4, 16));
  v = fmaxf(v, __shfl_xor(v, 8, 16));
  return v;
}
static __device__ __forceinline__ float rowsum16(float v) {
  v += __shfl_xor(v, 1, 16);
  v += __shfl_xor(v, 2, 16);
  v += __shfl_xor(v, 4, 16);
  v += __shfl_xor(v, 8, 16);
  return v;
}

// ---------------------------------------------------------------------------
// fp32 -> bf16 bulk convert (b128 load, b64 store)
// ---------------------------------------------------------------------------
__global__ __launch_bounds__(256)
void f32_to_bf16_kernel(const float* __restrict__ in, bf16_t* __restrict__ out, int n4) {
  int i = blockIdx.x * 256 + threadIdx.x;
  if (i >= n4) return;
  v4f x = ((const v4f*)in)[i];
  v4bf y;
  y[0] = (bf16_t)x[0]; y[1] = (bf16_t)x[1]; y[2] = (bf16_t)x[2]; y[3] = (bf16_t)x[3];
  ((v4bf*)out)[i] = y;
}

// ---------------------------------------------------------------------------
// Tiled bf16 WMMA GEMM:  C[M x N] = A[M x K](bf16) * B[K x N](fp32, cvt bf16)
// Block tile 128x128, 8 waves (2x4), each wave 64x32 (4x2 of 16x16).
// A tile (128 rows x 64B) is staged into LDS by the Tensor Data Mover
// (tensor_load_to_lds, TENSORcnt), with HW padding of 4 DWORDs every
// 16 DWORDs -> LDS row stride 40 bf16, matching the fragment addressing.
// MODE 0: store bf16 as [n>>7][m][n&127]   (per-head [S][D], for Q and K)
// MODE 1: store bf16 as [n>>7][n&127][m]   (per-head [D][S], V transposed)
// MODE 2: store fp32 row-major [m][n]      (final output)
// ---------------------------------------------------------------------------
template <int MODE>
__global__ __launch_bounds__(256)
void gemm_bf16_kernel(const bf16_t* __restrict__ A, const float* __restrict__ B,
                      void* __restrict__ Cout, int K, int N, int SDIM) {
  const int m0 = blockIdx.x * 128;
  const int n0 = blockIdx.y * 128;
  const int t    = threadIdx.x;
  const int lane = t & 31;
  const int wave = t >> 5;
  const int wm = wave >> 2;        // 0..1
  const int wn = wave & 3;         // 0..3
  const int l16 = lane & 15;
  const int hh  = lane >> 4;

  __shared__ bf16_t As[128 * 40];  // [row][k], stride 40 (TDM pad: 4dw / 16dw)
  __shared__ bf16_t Bs[128 * 40];  // [n][k],   stride 40 (transposed in LDS)

  v8f acc[4][2] = {};

  const int bkr = t >> 3, bcg = t & 7;      // B stage: 32 k-rows x 128 n

  // TDM descriptor invariants (D# group1, ISA 8.4):
  //  data_size=2B, pad_enable, pad_interval=3 (16 DWORDs), pad_amount=3 (4 DWORDs)
  //  tensor_dim0=K, tensor_dim1=2048 rows, tile = 32 x 128, dim0_stride=K
  const unsigned int g1w0 = (1u << 16) | (1u << 20) | (3u << 22) | (3u << 25);
  const unsigned int ldsA = (unsigned int)(size_t)(&As[0]);

  for (int k0 = 0; k0 < K; k0 += 32) {
    // ---- stage A tile via Tensor Data Mover (wave 0 issues the DMA) ----
    if (t < 32) {
      const unsigned long long ga =
          (unsigned long long)(size_t)(A + (size_t)m0 * K + k0);
      u32x4 g0;
      g0[0] = 1u;                                        // count=1, no gather
      g0[1] = ldsA;                                      // lds_addr
      g0[2] = (unsigned int)ga;                          // global_addr[31:0]
      g0[3] = (unsigned int)(ga >> 32) | 0x80000000u;    // addr[56:32] | type=2
      u32x8 g1;
      g1[0] = g1w0;
      g1[1] = ((unsigned int)K & 0xFFFFu) << 16;         // tensor_dim0 lo16
      g1[2] = ((unsigned int)K >> 16) | (2048u << 16);   // dim0 hi16 | dim1 lo16
      g1[3] = (32u << 16);                               // dim1 hi16 | tile_dim0=32
      g1[4] = 128u;                                      // tile_dim1=128, tile_dim2=0
      g1[5] = (unsigned int)K;                           // dim0_stride lo32
      g1[6] = 0u;
      g1[7] = 0u;
      asm volatile("tensor_load_to_lds %0, %1" :: "s"(g0), "s"(g1) : "memory");
    }
    // ---- stage B tile (fp32 coalesced b128 loads, bf16 transposed DS stores) ----
    {
      const float* src = B + (size_t)(k0 + bkr) * N + n0 + bcg * 16;
#pragma unroll
      for (int q = 0; q < 4; q++) {
        v4f b = *(const v4f*)(src + q * 4);
#pragma unroll
        for (int j = 0; j < 4; j++)
          Bs[(bcg * 16 + q * 4 + j) * 40 + bkr] = (bf16_t)b[j];
      }
    }
    if (k0 + 32 < K)
      __builtin_prefetch(B + (size_t)(k0 + 32 + bkr) * N + n0 + bcg * 16, 0, 1);

    __builtin_amdgcn_s_wait_tensorcnt(0);   // TDM tile landed in LDS
    __syncthreads();

    // ---- A fragments: 16x32, row = lane%16, k = (e<8 ? e+8*hh : 16+(e-8)+8*hh) ----
    v16bf afrag[4];
#pragma unroll
    for (int mi = 0; mi < 4; mi++) {
      const int row = wm * 64 + mi * 16 + l16;
      v8bf lo = *(const v8bf*)&As[row * 40 + 8 * hh];
      v8bf hi = *(const v8bf*)&As[row * 40 + 16 + 8 * hh];
      afrag[mi] = cat8(lo, hi);
    }
    // ---- B fragments: 32x16, col = lane%16, k = (e<8 ? e+8*hh : 16+(e-8)+8*hh) ----
#pragma unroll
    for (int ni = 0; ni < 2; ni++) {
      const int col = wn * 32 + ni * 16 + l16;
      v8bf lo = *(const v8bf*)&Bs[col * 40 + 8 * hh];
      v8bf hi = *(const v8bf*)&Bs[col * 40 + 16 + 8 * hh];
      v16bf bfrag = cat8(lo, hi);
#pragma unroll
      for (int mi = 0; mi < 4; mi++)
        acc[mi][ni] = WMMA_BF16(afrag[mi], bfrag, acc[mi][ni]);
    }
    __syncthreads();
  }

  // ---- epilogue: C layout lane l holds (M = r + 8*(l/16), N = l%16) per VGPR r ----
#pragma unroll
  for (int mi = 0; mi < 4; mi++) {
#pragma unroll
    for (int ni = 0; ni < 2; ni++) {
#pragma unroll
      for (int r = 0; r < 8; r++) {
        const int m = m0 + wm * 64 + mi * 16 + r + 8 * hh;
        const int n = n0 + wn * 32 + ni * 16 + l16;
        const float v = acc[mi][ni][r];
        if (MODE == 0) {
          ((bf16_t*)Cout)[((size_t)(n >> 7) * SDIM + m) * 128 + (n & 127)] = (bf16_t)v;
        } else if (MODE == 1) {
          ((bf16_t*)Cout)[((size_t)(n >> 7) * 128 + (n & 127)) * SDIM + m] = (bf16_t)v;
        } else {
          ((float*)Cout)[(size_t)m * N + n] = v;
        }
      }
    }
  }
}

// ---------------------------------------------------------------------------
// RoPE in-place on bf16 [heads][S][128]; thread owns pair (j, j+64), j<64
// ---------------------------------------------------------------------------
__global__ __launch_bounds__(256)
void rope_kernel(bf16_t* __restrict__ X, const int* __restrict__ pos_ids,
                 int heads, int S) {
  const size_t i = (size_t)blockIdx.x * 256 + threadIdx.x;
  const size_t total = (size_t)heads * S * 64;
  if (i >= total) return;
  const int j = (int)(i & 63);
  const size_t hs = i >> 6;               // head*S + s
  const int s = (int)(hs % (size_t)S);
  const float p = (float)pos_ids[s];
  // inv_freq = 10000^(-j/64) = exp(-j * ln(10000)/64)
  const float inv = __expf(-(float)j * 0.14391156831212787f);
  const float ang = p * inv;
  float sn, cs;
  __sincosf(ang, &sn, &cs);
  bf16_t* base = X + (hs << 7);
  const float x1 = (float)base[j];
  const float x2 = (float)base[j + 64];
  base[j]      = (bf16_t)(x1 * cs - x2 * sn);
  base[j + 64] = (bf16_t)(x2 * cs + x1 * sn);
}

// ---------------------------------------------------------------------------
// GQA causal flash attention, all-WMMA.
// Grid: (S/64 q-tiles, NH heads), 128 threads = 4 waves, wave owns 16 q rows.
// Q,K: [h][s][128] bf16 (RoPE'd); V: [kvh][128][s] bf16 (transposed);
// out: [s][NH*128] bf16 (A-matrix for the Wo GEMM).
// ---------------------------------------------------------------------------
__global__ __launch_bounds__(128)
void flash_attn_kernel(const bf16_t* __restrict__ Qb, const bf16_t* __restrict__ Kb,
                       const bf16_t* __restrict__ Vt, bf16_t* __restrict__ Ob) {
  const int S = 2048;
  const int h   = blockIdx.y;
  const int kvh = h >> 2;                  // NH/NKV = 4
  const int q0  = blockIdx.x * 64;
  const int t = threadIdx.x;
  const int wave = t >> 5, lane = t & 31;
  const int l16 = lane & 15, hh = lane >> 4;

  __shared__ bf16_t Pl[4 * 16 * 40];       // per-wave P staging, padded stride 40
  bf16_t* pw = &Pl[wave * 16 * 40];

  // Q A-fragments (16 rows x 128 d = 4 frags of 16x32)
  v16bf qf[4];
  {
    const bf16_t* qbase = Qb + ((size_t)h * S + (q0 + wave * 16 + l16)) * 128;
#pragma unroll
    for (int kk = 0; kk < 4; kk++) {
      v8bf lo = *(const v8bf*)(qbase + kk * 32 + 8 * hh);
      v8bf hi = *(const v8bf*)(qbase + kk * 32 + 16 + 8 * hh);
      qf[kk] = cat8(lo, hi);
    }
  }

  float mrow[8], lrow[8];
  v8f acc[8] = {};
#pragma unroll
  for (int r = 0; r < 8; r++) { mrow[r] = -1e30f; lrow[r] = 0.0f; }

  const float scale = 0.08838834764831845f;   // 1/sqrt(128)
  const int nch = (q0 + 64) >> 5;             // causal: chunks of 32 k-positions

  for (int kc = 0; kc < nch; kc++) {
    const int kbase = kc * 32;

    // ---- scores S = Q K^T : two 16-col tiles, contraction D=128 (4 WMMA each) ----
    v8f sc[2] = {};
#pragma unroll
    for (int tt = 0; tt < 2; tt++) {
      const int kpos = kbase + tt * 16 + l16;
      const bf16_t* kb = Kb + ((size_t)kvh * S + kpos) * 128;
#pragma unroll
      for (int kk = 0; kk < 4; kk++) {
        v8bf lo = *(const v8bf*)(kb + kk * 32 + 8 * hh);
        v8bf hi = *(const v8bf*)(kb + kk * 32 + 16 + 8 * hh);
        sc[tt] = WMMA_BF16(qf[kk], cat8(lo, hi), sc[tt]);
      }
    }

    // ---- online softmax (per-row state lives per-VGPR, width-16 reductions) ----
    float p0[8], p1[8], alpha[8];
#pragma unroll
    for (int r = 0; r < 8; r++) {
      const int row = q0 + wave * 16 + r + 8 * hh;
      float s0 = sc[0][r] * scale;
      float s1 = sc[1][r] * scale;
      if (kbase + l16 > row)      s0 = -1e30f;
      if (kbase + 16 + l16 > row) s1 = -1e30f;
      float mx = rowmax16(fmaxf(s0, s1));
      const float mnew = fmaxf(mrow[r], mx);
      const float e0 = __expf(s0 - mnew);
      const float e1 = __expf(s1 - mnew);
      p0[r] = e0; p1[r] = e1;
      const float sm = rowsum16(e0 + e1);
      alpha[r] = __expf(mrow[r] - mnew);
      lrow[r] = lrow[r] * alpha[r] + sm;
      mrow[r] = mnew;
    }
#pragma unroll
    for (int dt = 0; dt < 8; dt++)
#pragma unroll
      for (int r = 0; r < 8; r++) acc[dt][r] *= alpha[r];

    // ---- transpose P (C layout -> A layout) through per-wave LDS ----
#pragma unroll
    for (int r = 0; r < 8; r++) {
      const int row = r + 8 * hh;
      pw[row * 40 + l16]      = (bf16_t)p0[r];
      pw[row * 40 + 16 + l16] = (bf16_t)p1[r];
    }
    v8bf plo = *(const v8bf*)&pw[l16 * 40 + 8 * hh];
    v8bf phi = *(const v8bf*)&pw[l16 * 40 + 16 + 8 * hh];
    v16bf pf = cat8(plo, phi);

    // ---- O += P V : contraction 32 (chunk), 8 d-tiles; V^T gives b128 B-frags ----
#pragma unroll
    for (int dt = 0; dt < 8; dt++) {
      const bf16_t* vb = Vt + ((size_t)kvh * 128 + dt * 16 + l16) * S + kbase;
      v8bf lo = *(const v8bf*)(vb + 8 * hh);
      v8bf hi = *(const v8bf*)(vb + 16 + 8 * hh);
      acc[dt] = WMMA_BF16(pf, cat8(lo, hi), acc[dt]);
    }
  }

  // ---- normalize and store bf16 [s][NH*128] ----
#pragma unroll
  for (int r = 0; r < 8; r++) lrow[r] = 1.0f / lrow[r];
#pragma unroll
  for (int dt = 0; dt < 8; dt++)
#pragma unroll
    for (int r = 0; r < 8; r++) {
      const int row = q0 + wave * 16 + r + 8 * hh;
      const int col = h * 128 + dt * 16 + l16;
      Ob[(size_t)row * 4096 + col] = (bf16_t)(acc[dt][r] * lrow[r]);
    }
}

// ---------------------------------------------------------------------------
extern "C" void kernel_launch(void* const* d_in, const int* in_sizes, int n_in,
                              void* d_out, int out_size, void* d_ws, size_t ws_size,
                              hipStream_t stream) {
  (void)in_sizes; (void)n_in; (void)out_size; (void)ws_size;
  const int S = 2048, H = 4096, NH = 32, NKV = 8, D = 128;

  const float* hs  = (const float*)d_in[0];   // [1,S,H]
  const int*   pos = (const int*)d_in[1];     // [1,S]
  const float* Wq  = (const float*)d_in[2];   // [H, NH*D]
  const float* Wk  = (const float*)d_in[3];   // [H, NKV*D]
  const float* Wv  = (const float*)d_in[4];   // [H, NKV*D]
  const float* Wo  = (const float*)d_in[5];   // [NH*D, H]
  float* out = (float*)d_out;                 // [1,S,H] fp32

  // workspace carve-up (56 MB total)
  bf16_t* Xb = (bf16_t*)d_ws;                    // [S][H]        16 MB
  bf16_t* Qb = Xb + (size_t)S * H;               // [NH][S][D]    16 MB
  bf16_t* Kb = Qb + (size_t)NH * S * D;          // [NKV][S][D]    4 MB
  bf16_t* Vt = Kb + (size_t)NKV * S * D;         // [NKV][D][S]    4 MB
  bf16_t* Ab = Vt + (size_t)NKV * D * S;         // [S][NH*D]     16 MB

  // 1) hidden -> bf16
  {
    const int n4 = S * H / 4;
    f32_to_bf16_kernel<<<(n4 + 255) / 256, 256, 0, stream>>>(hs, Xb, n4);
  }
  // 2) QKV projections (WMMA GEMMs); V written pre-transposed
  gemm_bf16_kernel<0><<<dim3(S / 128, (NH * D) / 128), 256, 0, stream>>>(Xb, Wq, Qb, H, NH * D, S);
  gemm_bf16_kernel<0><<<dim3(S / 128, (NKV * D) / 128), 256, 0, stream>>>(Xb, Wk, Kb, H, NKV * D, S);
  gemm_bf16_kernel<1><<<dim3(S / 128, (NKV * D) / 128), 256, 0, stream>>>(Xb, Wv, Vt, H, NKV * D, S);
  // 3) RoPE on Q and K
  rope_kernel<<<(NH * S * 64 + 255) / 256, 256, 0, stream>>>(Qb, pos, NH, S);
  rope_kernel<<<(NKV * S * 64 + 255) / 256, 256, 0, stream>>>(Kb, pos, NKV, S);
  // 4) causal GQA flash attention
  flash_attn_kernel<<<dim3(S / 64, NH), 128, 0, stream>>>(Qb, Kb, Vt, Ab);
  // 5) output projection -> fp32
  gemm_bf16_kernel<2><<<dim3(S / 128, H / 128), 256, 0, stream>>>(Ab, Wo, out, NH * D, H, S);
}
